// MAPHead_63393717289548
// MI455X (gfx1250) — compile-verified, matrix operand order checked by reading, not run on hardware
//
#include <hip/hip_runtime.h>
#include <math.h>

#define N_   32
#define L_   4096
#define D_   768
#define H_   12
#define DH_  64
#define HP_  16      // heads padded to WMMA tile
#define MLP_ 3072
#define LSPLIT 8

typedef __attribute__((ext_vector_type(16))) __bf16 v16bf;
typedef __attribute__((ext_vector_type(8)))  float  f32x8;

union BFrag {
  unsigned short s[16];
  uint4 u[2];
  v16bf v;
};

// float -> bf16 raw bits via native hardware convert (RNE)
__device__ __forceinline__ unsigned short f2bfu(float f) {
  union { __bf16 b; unsigned short u; } c;
  c.b = (__bf16)f;
  return c.u;
}

// ---------------------------------------------------------------------------
// K1: q = (probe@wq + bq)/sqrt(DH); fold into wk -> wkq[d,h]; emit wkq directly
// in per-lane WMMA B-fragment layout (frag f = kt*32+lane, 16 bf16 each).
// Also lbias[h] = q[h,:]·bk[h,:].
// ---------------------------------------------------------------------------
__global__ void k1_prep(const float* __restrict__ probe,
                        const float* __restrict__ wq, const float* __restrict__ bq,
                        const float* __restrict__ wk, const float* __restrict__ bk,
                        unsigned short* __restrict__ wkqB, float* __restrict__ lbias) {
  __shared__ float qs[D_];
  int t = threadIdx.x;
  for (int idx = t; idx < D_; idx += 256) {
    float acc = bq[idx];
    for (int d = 0; d < D_; ++d) acc += probe[d] * wq[(size_t)d * D_ + idx];
    qs[idx] = acc * 0.125f;  // 1/sqrt(64)
  }
  __syncthreads();
  for (int i = t; i < 24 * 32 * 16; i += 256) {
    int j = i & 15;
    int f = i >> 4;
    int lane = f & 31;
    int kt = f >> 5;
    int col = lane & 15;                // N (head) column
    int kb = (lane >> 4) * 8;           // K sub-block per ISA A/B layout
    int K = kt * 32 + kb + (j < 8 ? j : j + 8);
    float v = 0.f;
    if (col < H_) {
      const float* wkp = wk + ((size_t)K * H_ + col) * DH_;
      const float* qp  = qs + col * DH_;
      for (int e = 0; e < DH_; ++e) v += wkp[e] * qp[e];
    }
    wkqB[i] = f2bfu(v);
  }
  if (t < HP_) {
    float v = 0.f;
    if (t < H_) for (int e = 0; e < DH_; ++e) v += bk[t * DH_ + e] * qs[t * DH_ + e];
    lbias[t] = v;
  }
}

// ---------------------------------------------------------------------------
// K2: logits[n,h,l] = x[n,l,:] @ wkq[:,h] + lbias[h]   (WMMA bf16, fp32 acc)
// grid: N * L/128 blocks, 8 waves/block, one 16-row l-tile per wave.
// ---------------------------------------------------------------------------
__global__ void k2_logits(const float* __restrict__ x,
                          const unsigned short* __restrict__ wkqB,
                          const float* __restrict__ lbias,
                          float* __restrict__ logits) {
  int bid = blockIdx.x;
  int n  = bid >> 5;
  int lt = bid & 31;
  int w    = threadIdx.x >> 5;
  int lane = threadIdx.x & 31;
  int m  = lane & 15;
  int kb = (lane >> 4) * 8;
  int lrow0 = lt * 128 + w * 16;
  const float* xrow = x + ((size_t)n * L_ + lrow0 + m) * D_;
  f32x8 acc = {};
  for (int kt = 0; kt < 24; ++kt) {
    int k0 = kt * 32 + kb;
    // prefetch ~2 k-tiles ahead (speculative; harmless at row end)
    __builtin_prefetch(xrow + k0 + 64, 0, 1);
    float4 a0 = *(const float4*)(xrow + k0);
    float4 a1 = *(const float4*)(xrow + k0 + 4);
    float4 a2 = *(const float4*)(xrow + k0 + 16);
    float4 a3 = *(const float4*)(xrow + k0 + 20);
    BFrag A;
    A.v[0]  = (__bf16)a0.x; A.v[1]  = (__bf16)a0.y; A.v[2]  = (__bf16)a0.z; A.v[3]  = (__bf16)a0.w;
    A.v[4]  = (__bf16)a1.x; A.v[5]  = (__bf16)a1.y; A.v[6]  = (__bf16)a1.z; A.v[7]  = (__bf16)a1.w;
    A.v[8]  = (__bf16)a2.x; A.v[9]  = (__bf16)a2.y; A.v[10] = (__bf16)a2.z; A.v[11] = (__bf16)a2.w;
    A.v[12] = (__bf16)a3.x; A.v[13] = (__bf16)a3.y; A.v[14] = (__bf16)a3.z; A.v[15] = (__bf16)a3.w;
    BFrag B;
    const uint4* bp = (const uint4*)(wkqB + ((size_t)kt * 32 + lane) * 16);
    B.u[0] = bp[0]; B.u[1] = bp[1];
    acc = __builtin_amdgcn_wmma_f32_16x16x32_bf16(false, A.v, false, B.v,
                                                  (short)0, acc, false, false);
  }
  int h = lane & 15;          // C/D: lane holds column N, VGPR r = row r(+8)
  if (h < H_) {
    float lb = lbias[h];
    float* dst = logits + ((size_t)n * H_ + h) * L_ + lrow0 + ((lane >> 4) * 8);
    float4 o0 = { acc[0] + lb, acc[1] + lb, acc[2] + lb, acc[3] + lb };
    float4 o1 = { acc[4] + lb, acc[5] + lb, acc[6] + lb, acc[7] + lb };
    *(float4*)dst = o0;
    *(float4*)(dst + 4) = o1;
  }
}

// ---------------------------------------------------------------------------
// K3: softmax over L per (n,h); write bf16 attn [N][HP][L], rows h>=12 zeroed.
// ---------------------------------------------------------------------------
__global__ void k3_softmax(const float* __restrict__ logits,
                           unsigned short* __restrict__ attn) {
  __shared__ float red[256];
  int bid = blockIdx.x;
  int n = bid >> 4;
  int h = bid & 15;
  int t = threadIdx.x;
  unsigned short* arow = attn + ((size_t)n * HP_ + h) * L_;
  if (h >= H_) {
    for (int i = t; i < L_; i += 256) arow[i] = 0;
    return;
  }
  const float* lrow = logits + ((size_t)n * H_ + h) * L_;
  float mx = -1e30f;
  for (int i = t; i < L_; i += 256) mx = fmaxf(mx, lrow[i]);
  red[t] = mx; __syncthreads();
  for (int s = 128; s > 0; s >>= 1) { if (t < s) red[t] = fmaxf(red[t], red[t + s]); __syncthreads(); }
  mx = red[0]; __syncthreads();
  float sm = 0.f;
  for (int i = t; i < L_; i += 256) sm += __expf(lrow[i] - mx);
  red[t] = sm; __syncthreads();
  for (int s = 128; s > 0; s >>= 1) { if (t < s) red[t] += red[t + s]; __syncthreads(); }
  float inv = 1.0f / red[0];
  for (int i = t; i < L_; i += 256) arow[i] = f2bfu(__expf(lrow[i] - mx) * inv);
}

// ---------------------------------------------------------------------------
// K4: xbar_part[ls][n][h][d] = sum over l-slice of attn[n,h,l]*x[n,l,d] (WMMA)
// grid: N*LSPLIT blocks, 8 waves, 6 d-tiles of 16 per wave (48*16=768).
// ---------------------------------------------------------------------------
__global__ void k4_xbar(const float* __restrict__ x,
                        const unsigned short* __restrict__ attn,
                        float* __restrict__ xpart) {
  int bid = blockIdx.x;
  int n  = bid >> 3;
  int ls = bid & 7;
  int w    = threadIdx.x >> 5;
  int lane = threadIdx.x & 31;
  int m  = lane & 15;
  int kb = (lane >> 4) * 8;
  const unsigned short* arow = attn + ((size_t)n * HP_ + m) * L_ + ls * 512 + kb;
  f32x8 zero = {};
  f32x8 acc[6];
  for (int i = 0; i < 6; ++i) acc[i] = zero;
  for (int lt = 0; lt < 16; ++lt) {
    int l0 = lt * 32;
    BFrag A;
    A.u[0] = *(const uint4*)(arow + l0);
    A.u[1] = *(const uint4*)(arow + l0 + 16);
    int lg = ls * 512 + l0;
    for (int i = 0; i < 6; ++i) {
      int d = (w * 6 + i) * 16 + m;
      const float* xb = x + ((size_t)n * L_ + lg + kb) * D_ + d;
      BFrag B;
#pragma unroll
      for (int j = 0; j < 16; ++j) {
        int K = (j < 8 ? j : j + 8);
        B.v[j] = (__bf16)xb[(size_t)K * D_];
      }
      acc[i] = __builtin_amdgcn_wmma_f32_16x16x32_bf16(false, A.v, false, B.v,
                                                       (short)0, acc[i], false, false);
    }
  }
  int rb = (lane >> 4) * 8;
  for (int i = 0; i < 6; ++i) {
    int d = (w * 6 + i) * 16 + m;
    float* dst = xpart + (((size_t)ls * N_ + n) * HP_ + rb) * D_ + d;
    for (int r = 0; r < 8; ++r) dst[(size_t)r * D_] = acc[i][r];
  }
}

__global__ void k4b_reduce(const float* __restrict__ xpart, float* __restrict__ xbar) {
  int idx = blockIdx.x * 256 + threadIdx.x;   // < N*HP*D
  float s = 0.f;
  for (int p = 0; p < LSPLIT; ++p) s += xpart[(size_t)p * (N_ * HP_ * D_) + idx];
  xbar[idx] = s;
}

// ---------------------------------------------------------------------------
// K5: per-sample head mix: o = xbar@wv+bv; xa = o@wo+bo; y = LayerNorm(xa)
// ---------------------------------------------------------------------------
__global__ void k5_head(const float* __restrict__ xbar,
                        const float* __restrict__ wv, const float* __restrict__ bv,
                        const float* __restrict__ wo, const float* __restrict__ bo,
                        const float* __restrict__ lns, const float* __restrict__ lnb,
                        float* __restrict__ xa, float* __restrict__ y) {
  __shared__ float oL[D_];
  __shared__ float xaL[D_];
  __shared__ float red[256];
  int n = blockIdx.x;
  int t = threadIdx.x;
  const float* xb = xbar + (size_t)n * HP_ * D_;
  for (int idx = t; idx < H_ * DH_; idx += 256) {
    int h = idx >> 6, e = idx & 63;
    float acc = bv[idx];
    const float* xr = xb + (size_t)h * D_;
    for (int d = 0; d < D_; ++d) acc += xr[d] * wv[((size_t)d * H_ + h) * DH_ + e];
    oL[idx] = acc;
  }
  __syncthreads();
  for (int idx = t; idx < D_; idx += 256) {
    float acc = bo[idx];
    for (int k = 0; k < H_ * DH_; ++k) acc += oL[k] * wo[(size_t)k * D_ + idx];
    xaL[idx] = acc;
    xa[(size_t)n * D_ + idx] = acc;
  }
  __syncthreads();
  float s = 0.f, s2 = 0.f;
  for (int idx = t; idx < D_; idx += 256) { float v = xaL[idx]; s += v; s2 += v * v; }
  red[t] = s; __syncthreads();
  for (int st = 128; st > 0; st >>= 1) { if (t < st) red[t] += red[t + st]; __syncthreads(); }
  float mu = red[0] / D_; __syncthreads();
  red[t] = s2; __syncthreads();
  for (int st = 128; st > 0; st >>= 1) { if (t < st) red[t] += red[t + st]; __syncthreads(); }
  float var = red[0] / D_ - mu * mu;
  float rstd = rsqrtf(var + 1e-6f);
  for (int idx = t; idx < D_; idx += 256)
    y[(size_t)n * D_ + idx] = (xaL[idx] - mu) * rstd * lns[idx] + lnb[idx];
}

// ---------------------------------------------------------------------------
// K6/K7: MLP, weights read exactly once (column-parallel), activations via LDS
// ---------------------------------------------------------------------------
__global__ void k6_mlp1(const float* __restrict__ y, const float* __restrict__ w1,
                        const float* __restrict__ b1, float* __restrict__ h1) {
  __shared__ float yL[N_ * 64];
  int t = threadIdx.x;
  int c = blockIdx.x * 256 + t;
  float acc[N_];
  float bb = b1[c];
  for (int n = 0; n < N_; ++n) acc[n] = bb;
  for (int dt = 0; dt < D_ / 64; ++dt) {
    __syncthreads();
    for (int i = t * 8; i < t * 8 + 8; ++i) {
      int n = i >> 6, d = i & 63;
      yL[i] = y[(size_t)n * D_ + dt * 64 + d];
    }
    __syncthreads();
    for (int d = 0; d < 64; ++d) {
      float wv_ = w1[(size_t)(dt * 64 + d) * MLP_ + c];
#pragma unroll
      for (int n = 0; n < N_; ++n) acc[n] = fmaf(yL[n * 64 + d], wv_, acc[n]);
    }
  }
  for (int n = 0; n < N_; ++n) {
    float a = acc[n];
    float g = 0.5f * a * (1.0f + tanhf(0.7978845608f * (a + 0.044715f * a * a * a)));
    h1[(size_t)n * MLP_ + c] = g;
  }
}

__global__ void k7_mlp2(const float* __restrict__ h1, const float* __restrict__ w2,
                        const float* __restrict__ b2, const float* __restrict__ xa,
                        float* __restrict__ out) {
  __shared__ float hL[N_ * 64];
  int t = threadIdx.x;
  int c = blockIdx.x * 256 + t;
  float acc[N_];
  float bb = b2[c];
  for (int n = 0; n < N_; ++n) acc[n] = bb;
  for (int kt = 0; kt < MLP_ / 64; ++kt) {
    __syncthreads();
    for (int i = t * 8; i < t * 8 + 8; ++i) {
      int n = i >> 6, k = i & 63;
      hL[i] = h1[(size_t)n * MLP_ + kt * 64 + k];
    }
    __syncthreads();
    for (int k = 0; k < 64; ++k) {
      float wv_ = w2[(size_t)(kt * 64 + k) * D_ + c];
#pragma unroll
      for (int n = 0; n < N_; ++n) acc[n] = fmaf(hL[n * 64 + k], wv_, acc[n]);
    }
  }
  for (int n = 0; n < N_; ++n)
    out[(size_t)n * D_ + c] = xa[(size_t)n * D_ + c] + acc[n];
}

// ---------------------------------------------------------------------------
extern "C" void kernel_launch(void* const* d_in, const int* in_sizes, int n_in,
                              void* d_out, int out_size, void* d_ws, size_t ws_size,
                              hipStream_t stream) {
  (void)in_sizes; (void)n_in; (void)out_size; (void)ws_size;
  const float* x     = (const float*)d_in[0];
  const float* probe = (const float*)d_in[1];
  const float* wq    = (const float*)d_in[2];
  const float* bq    = (const float*)d_in[3];
  const float* wk    = (const float*)d_in[4];
  const float* bk    = (const float*)d_in[5];
  const float* wv    = (const float*)d_in[6];
  const float* bv    = (const float*)d_in[7];
  const float* wo    = (const float*)d_in[8];
  const float* bo    = (const float*)d_in[9];
  const float* lns   = (const float*)d_in[10];
  const float* lnb   = (const float*)d_in[11];
  const float* w1    = (const float*)d_in[12];
  const float* b1    = (const float*)d_in[13];
  const float* w2    = (const float*)d_in[14];
  const float* b2    = (const float*)d_in[15];

  char* p = (char*)d_ws;
  auto alloc = [&](size_t bytes) -> char* {
    char* r = p; p += (bytes + 255) & ~(size_t)255; return r;
  };
  unsigned short* wkqB  = (unsigned short*)alloc((size_t)24 * 32 * 16 * 2);
  float*          lbias = (float*)alloc(HP_ * 4);
  float*          logit = (float*)alloc((size_t)N_ * H_ * L_ * 4);
  unsigned short* attn  = (unsigned short*)alloc((size_t)N_ * HP_ * L_ * 2);
  float*          xpart = (float*)alloc((size_t)LSPLIT * N_ * HP_ * D_ * 4);
  float*          xbar  = (float*)alloc((size_t)N_ * HP_ * D_ * 4);
  float*          xa    = (float*)alloc((size_t)N_ * D_ * 4);
  float*          y     = (float*)alloc((size_t)N_ * D_ * 4);
  float*          h1    = (float*)alloc((size_t)N_ * MLP_ * 4);
  float*          out   = (float*)d_out;

  k1_prep   <<<1,                 256, 0, stream>>>(probe, wq, bq, wk, bk, wkqB, lbias);
  k2_logits <<<N_ * (L_ / 128),   256, 0, stream>>>(x, wkqB, lbias, logit);
  k3_softmax<<<N_ * HP_,          256, 0, stream>>>(logit, attn);
  k4_xbar   <<<N_ * LSPLIT,       256, 0, stream>>>(x, attn, xpart);
  k4b_reduce<<<(N_ * HP_ * D_) / 256, 256, 0, stream>>>(xpart, xbar);
  k5_head   <<<N_,                256, 0, stream>>>(xbar, wv, bv, wo, bo, lns, lnb, xa, y);
  k6_mlp1   <<<MLP_ / 256,        256, 0, stream>>>(y, w1, b1, h1);
  k7_mlp2   <<<D_ / 256,          256, 0, stream>>>(h1, w2, b2, xa, out);
}
